// Net_50122268344779
// MI455X (gfx1250) — compile-verified, matrix operand order checked by reading, not run on hardware
//
#include <hip/hip_runtime.h>
#include <hip/hip_bf16.h>

// ---------------------------------------------------------------------------
// GCN forward for MI455X (gfx1250):
//   bf16 pre-conversion (L2-resident A) + TDM-staged, double-buffered WMMA GEMM
//   with software-pipelined B operand + atomic edge scatter.
//   Fallback f32-staging GEMM if workspace is small.
// ---------------------------------------------------------------------------

#define N_NODES 50000
#define N_EDGES 400000
#define IN_FEATS 1433
#define HID 384
#define EMB 256
#define N_CLS 7

#define MP 50048          // N_NODES padded to multiple of 128 (391 * 128)
#define KP1 1440          // IN_FEATS padded to multiple of 32

typedef __attribute__((ext_vector_type(16))) __bf16 v16bf;
typedef __attribute__((ext_vector_type(8)))  float  v8f;
typedef __attribute__((ext_vector_type(4)))  unsigned int u32x4;
typedef __attribute__((ext_vector_type(8)))  int  i32x8;
typedef __attribute__((ext_vector_type(4)))  int  i32x4;
typedef unsigned short ushort_t;

__device__ __forceinline__ unsigned short f2bf(float f) {
  unsigned int u = __float_as_uint(f);
  u += 0x7FFFu + ((u >> 16) & 1u);          // round-to-nearest-even
  return (unsigned short)(u >> 16);
}

// ---------------------------------------------------------------------------
// zero fill / degrees / inverse sqrt
// ---------------------------------------------------------------------------
__global__ void zero_f32(float* __restrict__ p, int n) {
  int i = blockIdx.x * blockDim.x + threadIdx.x;
  if (i < n) p[i] = 0.0f;
}

__global__ void degree_kernel(const int* __restrict__ src, const int* __restrict__ dst,
                              float* __restrict__ deg_out, float* __restrict__ deg_in, int E) {
  int e = blockIdx.x * blockDim.x + threadIdx.x;
  if (e < E) {
    atomicAdd(&deg_out[src[e]], 1.0f);
    atomicAdd(&deg_in[dst[e]], 1.0f);
  }
}

__global__ void invsqrt_kernel(const float* __restrict__ deg_out, const float* __restrict__ deg_in,
                               float* __restrict__ inv_s, float* __restrict__ inv_d, int N) {
  int i = blockIdx.x * blockDim.x + threadIdx.x;
  if (i < N) {
    inv_s[i] = rsqrtf(fmaxf(deg_out[i], 1.0f));
    inv_d[i] = rsqrtf(fmaxf(deg_in[i], 1.0f));
  }
}

// ---------------------------------------------------------------------------
// f32 -> bf16 conversion with row/col zero padding.
// ---------------------------------------------------------------------------
__global__ void cvt_pad_bf16(const float* __restrict__ src, ushort_t* __restrict__ dst,
                             int validRows, int K, int Kp) {
  int m = blockIdx.x;
  bool rowOk = (m < validRows);
  for (int k = threadIdx.x; k < Kp; k += blockDim.x) {
    float v = (rowOk && k < K) ? src[(size_t)m * K + k] : 0.0f;
    dst[(size_t)m * Kp + k] = f2bf(v);
  }
}

// Layer-1 epilogue fused into conversion: h1 = relu(agg1*inv_d[m] + b1[k]) -> bf16
__global__ void cvt_h1_bf16(const float* __restrict__ agg1, const float* __restrict__ inv_d,
                            const float* __restrict__ b1, ushort_t* __restrict__ dst, int M) {
  int m = blockIdx.x;                     // grid.x == MP
  bool rowOk = (m < M);
  float s = rowOk ? inv_d[m] : 0.0f;
  for (int k = threadIdx.x; k < HID; k += blockDim.x) {
    float v = 0.0f;
    if (rowOk) v = fmaxf(fmaf(agg1[(size_t)m * HID + k], s, b1[k]), 0.0f);
    dst[(size_t)m * HID + k] = f2bf(v);
  }
}

// ---------------------------------------------------------------------------
// TDM helper: DMA a 128-row x 32-col bf16 tile (row stride Kp elements) from
// global into LDS. D# per CDNA5 ISA §8.3-8.5. Issued by one wave per block.
// 6-arg builtin on this toolchain: (u32x4, i32x8, i32x4, i32x4, i32x8, cpol).
// ---------------------------------------------------------------------------
__device__ __forceinline__ void tdm_load_a_tile(const ushort_t* gsrc, unsigned lds_byte_addr,
                                                int Kp) {
  unsigned long long ga = (unsigned long long)(uintptr_t)gsrc;
  // Group 0: count=1 | lds_addr | global_addr | type=2 ("image")
  u32x4 g0 = { 1u,
               lds_byte_addr,
               (unsigned)ga,
               (unsigned)((ga >> 32) & 0x01FFFFFFu) | 0x80000000u };
  // Group 1: data_size=2B; tensor 32x128 (tile-exact; padded buffers keep all
  // accesses in-bounds); tile 32x128; row stride = Kp elements.
  i32x8 g1 = { (int)(1u << 16),                     // data_size=1 -> 2 bytes
               (int)((32u & 0xFFFFu) << 16),        // tensor_dim0[15:0]
               (int)((128u & 0xFFFFu) << 16),       // dim0[31:16]=0 | tensor_dim1[15:0]
               (int)(32u << 16),                    // dim1[31:16]=0 | tile_dim0=32
               (int)128u,                           // tile_dim1=128, tile_dim2=0
               (int)(unsigned)Kp,                   // tensor_dim0_stride[31:0]
               0, 0 };
  i32x4 gz4 = { 0, 0, 0, 0 };
  i32x8 gz8 = { 0, 0, 0, 0, 0, 0, 0, 0 };
  __builtin_amdgcn_tensor_load_to_lds(g0, g1, gz4, gz4, gz8, 0);
}

// ---------------------------------------------------------------------------
// Path A GEMM: A (bf16, [MP][Kp] padded) @ W (bf16, [Kp][Nn]) -> out f32,
// scaled per-row by post_scale.
//  - A tiles staged by TDM (async, double-buffered LDS)
//  - B tiles software-pipelined 2 steps deep: global load for step kt+2 issued
//    at step kt; LDS store of the (already-arrived) step kt+1 data at step kt.
//  - All fragment ds_loads batched ahead of the 4 back-to-back WMMAs.
// ---------------------------------------------------------------------------
__global__ __launch_bounds__(256) void gemm_wmma_bf16_tdm(
    const ushort_t* __restrict__ A, const ushort_t* __restrict__ W,
    float* __restrict__ out, const float* __restrict__ post_scale,
    int M, int Kp, int Nn)
{
  __shared__ __align__(16) ushort_t lds_a[2][128 * 32];  // [buf][row*32+k]
  __shared__ __align__(16) ushort_t lds_b[2][64 * 32];   // [buf][n*32+k]

  const int tid   = threadIdx.x;
  const int mBase = blockIdx.x * 128;
  const int nBase = blockIdx.y * 64;
  const int wave  = tid >> 5;
  const int lane  = tid & 31;
  const int hi    = lane >> 4;
  const int r     = lane & 15;
  const int waveM = wave * 16;

  const ushort_t* Atile = A + (size_t)mBase * Kp;

  v8f acc[4] = {};

  // B staging geometry: 256 threads cover the 32x64 tile in 8-element rows.
  const int bk  = tid >> 3;        // 0..31  (k within tile)
  const int bn8 = tid & 7;         // 0..7   (group of 8 n values)
  const ushort_t* Bsrc = W + (size_t)bk * Nn + nBase + bn8 * 8;

  auto load_b = [&](int kt) -> uint4 {
    return *reinterpret_cast<const uint4*>(Bsrc + ((size_t)kt << 5) * Nn);
  };
  auto store_b = [&](uint4 v, int buf) {
    union { uint4 q; ushort_t s[8]; } u;
    u.q = v;
    #pragma unroll
    for (int j = 0; j < 8; ++j) lds_b[buf][(bn8 * 8 + j) * 32 + bk] = u.s[j];
  };

  const int kIters = Kp >> 5;

  // ---- prologue: B(0) into LDS buf0, A(0) via TDM, prefetch B(1) ----
  uint4 breg = load_b(0);
  if (wave == 0)
    tdm_load_a_tile(Atile, (unsigned)(uintptr_t)&lds_a[0][0], Kp);
  store_b(breg, 0);
  if (kIters > 1) breg = load_b(1);          // arrives during step 0 compute

  for (int kt = 0; kt < kIters; ++kt) {
    const int buf = kt & 1;
    if (wave == 0) __builtin_amdgcn_s_wait_tensorcnt(0);  // A(kt) in LDS
    __syncthreads();                                      // + B(kt) visible

    // Stage step kt+1 (register data already in flight) and prefetch kt+2.
    if (kt + 1 < kIters) {
      store_b(breg, buf ^ 1);
      if (wave == 0)
        tdm_load_a_tile(Atile + ((size_t)(kt + 1) << 5),
                        (unsigned)(uintptr_t)&lds_a[buf ^ 1][0], Kp);
      if (kt + 2 < kIters) breg = load_b(kt + 2);
    }

    // Batched fragment loads (ISA 16-bit operand layout):
    //  lane group 0: v[0..7]=K 0..7,  v[8..15]=K 16..23
    //  lane group 1: v[0..7]=K 8..15, v[8..15]=K 24..31
    union Frag { uint4 q[2]; v16bf v; };
    Frag fa, fb[4];
    const ushort_t* la = &lds_a[buf][(waveM + r) * 32 + hi * 8];
    fa.q[0] = *reinterpret_cast<const uint4*>(la);
    fa.q[1] = *reinterpret_cast<const uint4*>(la + 16);
    #pragma unroll
    for (int nt = 0; nt < 4; ++nt) {
      const ushort_t* lb = &lds_b[buf][(nt * 16 + r) * 32 + hi * 8];
      fb[nt].q[0] = *reinterpret_cast<const uint4*>(lb);
      fb[nt].q[1] = *reinterpret_cast<const uint4*>(lb + 16);
    }
    #pragma unroll
    for (int nt = 0; nt < 4; ++nt) {
      acc[nt] = __builtin_amdgcn_wmma_f32_16x16x32_bf16(
          false, fa.v, false, fb[nt].v, (short)0, acc[nt], false, false);
    }
    __syncthreads();   // reads done before this buffer is restaged
  }

  // Epilogue: C layout VGPR j -> row (j + 8*hi), col = lane r.
  #pragma unroll
  for (int nt = 0; nt < 4; ++nt) {
    int col = nBase + nt * 16 + r;
    #pragma unroll
    for (int j = 0; j < 8; ++j) {
      int m = mBase + waveM + hi * 8 + j;
      if (m < M) out[(size_t)m * Nn + col] = acc[nt][j] * post_scale[m];
    }
  }
}

// ---------------------------------------------------------------------------
// Path B (fallback, small workspace): f32-source GEMM with in-kernel bf16
// conversion.
// ---------------------------------------------------------------------------
__global__ __launch_bounds__(256) void gemm_wmma_f32src(
    const float* __restrict__ A, const float* __restrict__ W, float* __restrict__ out,
    const float* __restrict__ pre_scale, const float* __restrict__ pre_bias, int do_relu,
    const float* __restrict__ post_scale, int M, int K, int Nn)
{
  __shared__ __align__(16) ushort_t lds_a[128 * 32];
  __shared__ __align__(16) ushort_t lds_b[64 * 32];

  const int tid   = threadIdx.x;
  const int mBase = blockIdx.x * 128;
  const int nBase = blockIdx.y * 64;
  const int wave  = tid >> 5;
  const int lane  = tid & 31;
  const int hi    = lane >> 4;
  const int r     = lane & 15;
  const int waveM = wave * 16;
  const bool xform = (pre_scale != nullptr);

  v8f acc[4] = {};

  const int kIters = (K + 31) >> 5;
  for (int kt = 0; kt < kIters; ++kt) {
    const int k0 = kt << 5;
    #pragma unroll
    for (int i = 0; i < 16; ++i) {
      int idx = tid + i * 256;
      int row = idx >> 5, kk = idx & 31;
      int gm = mBase + row, gk = k0 + kk;
      float v = 0.0f;
      if (gm < M && gk < K) {
        v = A[(size_t)gm * K + gk];
        if (xform) {
          v = fmaf(v, pre_scale[gm], pre_bias[gk]);
          if (do_relu) v = fmaxf(v, 0.0f);
        }
      }
      lds_a[idx] = f2bf(v);
    }
    #pragma unroll
    for (int i = 0; i < 2; ++i) {
      int idx = tid + i * 256;
      int kk = idx >> 4, n4 = idx & 15;
      int gk = k0 + kk, gn = nBase + n4 * 4;
      float4 v = make_float4(0.f, 0.f, 0.f, 0.f);
      if (gk < K) v = *reinterpret_cast<const float4*>(&W[(size_t)gk * Nn + gn]);
      lds_b[(n4 * 4 + 0) * 32 + kk] = f2bf(v.x);
      lds_b[(n4 * 4 + 1) * 32 + kk] = f2bf(v.y);
      lds_b[(n4 * 4 + 2) * 32 + kk] = f2bf(v.z);
      lds_b[(n4 * 4 + 3) * 32 + kk] = f2bf(v.w);
    }
    __syncthreads();

    union Frag { uint4 q[2]; v16bf v; };
    Frag fa, fb[4];
    const int aoff = (waveM + r) * 32 + hi * 8;
    fa.q[0] = *reinterpret_cast<const uint4*>(&lds_a[aoff]);
    fa.q[1] = *reinterpret_cast<const uint4*>(&lds_a[aoff + 16]);
    #pragma unroll
    for (int nt = 0; nt < 4; ++nt) {
      const int boff = (nt * 16 + r) * 32 + hi * 8;
      fb[nt].q[0] = *reinterpret_cast<const uint4*>(&lds_b[boff]);
      fb[nt].q[1] = *reinterpret_cast<const uint4*>(&lds_b[boff + 16]);
    }
    #pragma unroll
    for (int nt = 0; nt < 4; ++nt) {
      acc[nt] = __builtin_amdgcn_wmma_f32_16x16x32_bf16(
          false, fa.v, false, fb[nt].v, (short)0, acc[nt], false, false);
    }
    __syncthreads();
  }

  #pragma unroll
  for (int nt = 0; nt < 4; ++nt) {
    int col = nBase + nt * 16 + r;
    #pragma unroll
    for (int j = 0; j < 8; ++j) {
      int m = mBase + waveM + hi * 8 + j;
      if (m < M) out[(size_t)m * Nn + col] = acc[nt][j] * post_scale[m];
    }
  }
}

// ---------------------------------------------------------------------------
// Edge scatter: agg[dst[e]][f] += h[src[e]][f]   (coalesced in f)
// ---------------------------------------------------------------------------
__global__ void scatter_add(const float* __restrict__ h, const int* __restrict__ src,
                            const int* __restrict__ dst, float* __restrict__ agg,
                            int E, int F) {
  int idx = blockIdx.x * blockDim.x + threadIdx.x;
  if (idx >= E * F) return;
  int e = idx / F;
  int f = idx - e * F;
  atomicAdd(&agg[(size_t)dst[e] * F + f], h[(size_t)src[e] * F + f]);
}

// ---------------------------------------------------------------------------
// Classifier: out[m][c] = bc[c] + sum_f (agg2[m][f]*inv_d[m] + b2[f]) * Wc[f][c]
// ---------------------------------------------------------------------------
__global__ __launch_bounds__(256) void classify_kernel(
    const float* __restrict__ agg2, const float* __restrict__ inv_d,
    const float* __restrict__ b2, const float* __restrict__ Wc,
    const float* __restrict__ bc, float* __restrict__ out, int M) {
  int gwave = (blockIdx.x * blockDim.x + threadIdx.x) >> 5;
  int lane  = threadIdx.x & 31;
  if (gwave >= M) return;

  const float* row = agg2 + (size_t)gwave * EMB;
  float s = inv_d[gwave];
  float acc[N_CLS] = {};
  for (int f = lane; f < EMB; f += 32) {
    float h = fmaf(row[f], s, b2[f]);
    #pragma unroll
    for (int c = 0; c < N_CLS; ++c) acc[c] = fmaf(h, Wc[f * N_CLS + c], acc[c]);
  }
  #pragma unroll
  for (int c = 0; c < N_CLS; ++c) {
    #pragma unroll
    for (int off = 16; off >= 1; off >>= 1) acc[c] += __shfl_xor(acc[c], off, 32);
  }
  if (lane == 0) {
    #pragma unroll
    for (int c = 0; c < N_CLS; ++c) out[(size_t)gwave * N_CLS + c] = acc[c] + bc[c];
  }
}

// ---------------------------------------------------------------------------
extern "C" void kernel_launch(void* const* d_in, const int* in_sizes, int n_in,
                              void* d_out, int out_size, void* d_ws, size_t ws_size,
                              hipStream_t stream) {
  const float* features = (const float*)d_in[0];
  const int*   esrc     = (const int*)d_in[1];
  const int*   edst     = (const int*)d_in[2];
  const float* W1       = (const float*)d_in[3];
  const float* b1       = (const float*)d_in[4];
  const float* W2       = (const float*)d_in[5];
  const float* b2       = (const float*)d_in[6];
  const float* Wc       = (const float*)d_in[7];
  const float* bc       = (const float*)d_in[8];
  float* out = (float*)d_out;

  const int B = 256;
  const int NPAD = 50016;                         // per-array float padding (16B multiples)

  float* ws = (float*)d_ws;
  size_t off = 0;
  float* deg_out = ws + off; off += NPAD;
  float* deg_in  = ws + off; off += NPAD;
  float* inv_s   = ws + off; off += NPAD;
  float* inv_d   = ws + off; off += NPAD;
  float* h1pre   = ws + off; off += (size_t)N_NODES * HID;
  float* agg1    = ws + off; off += (size_t)N_NODES * HID;
  float* h2pre   = ws + off; off += (size_t)N_NODES * EMB;
  float* agg2    = ws + off; off += (size_t)N_NODES * EMB;

  // bf16 region (path A only)
  ushort_t* u16base = (ushort_t*)(ws + off);
  size_t uoff = 0;
  ushort_t* featbf = u16base + uoff; uoff += (size_t)MP * KP1;       // 50048 x 1440
  ushort_t* w1bf   = u16base + uoff; uoff += (size_t)KP1 * HID;      // 1440 x 384
  ushort_t* h1bf   = u16base + uoff; uoff += (size_t)MP * HID;       // 50048 x 384
  ushort_t* w2bf   = u16base + uoff; uoff += (size_t)HID * EMB;      // 384 x 256
  const size_t needA = off * sizeof(float) + uoff * sizeof(ushort_t);
  const bool pathA = (ws_size >= needA);

  // Zero accumulation buffers every call (graph-replay determinism).
  zero_f32<<<(2 * NPAD + B - 1) / B, B, 0, stream>>>(deg_out, 2 * NPAD);
  zero_f32<<<(N_NODES * HID + B - 1) / B, B, 0, stream>>>(agg1, N_NODES * HID);
  zero_f32<<<(N_NODES * EMB + B - 1) / B, B, 0, stream>>>(agg2, N_NODES * EMB);

  degree_kernel<<<(N_EDGES + B - 1) / B, B, 0, stream>>>(esrc, edst, deg_out, deg_in, N_EDGES);
  invsqrt_kernel<<<(N_NODES + B - 1) / B, B, 0, stream>>>(deg_out, deg_in, inv_s, inv_d, N_NODES);

  dim3 grid1((N_NODES + 127) / 128, HID / 64);
  dim3 grid2((N_NODES + 127) / 128, EMB / 64);

  if (pathA) {
    // Pre-convert to padded bf16 (features become L2-resident: 144 MB < 192 MB).
    cvt_pad_bf16<<<MP, B, 0, stream>>>(features, featbf, N_NODES, IN_FEATS, KP1);
    cvt_pad_bf16<<<KP1, B, 0, stream>>>(W1, w1bf, IN_FEATS, HID, HID);

    gemm_wmma_bf16_tdm<<<grid1, B, 0, stream>>>(featbf, w1bf, h1pre, inv_s,
                                                N_NODES, KP1, HID);
    scatter_add<<<(N_EDGES * HID + B - 1) / B, B, 0, stream>>>(h1pre, esrc, edst, agg1,
                                                               N_EDGES, HID);

    cvt_h1_bf16<<<MP, B, 0, stream>>>(agg1, inv_d, b1, h1bf, N_NODES);
    cvt_pad_bf16<<<HID, B, 0, stream>>>(W2, w2bf, HID, EMB, EMB);

    gemm_wmma_bf16_tdm<<<grid2, B, 0, stream>>>(h1bf, w2bf, h2pre, inv_s,
                                                N_NODES, HID, EMB);
  } else {
    gemm_wmma_f32src<<<grid1, B, 0, stream>>>(features, W1, h1pre,
                                              nullptr, nullptr, 0,
                                              inv_s, N_NODES, IN_FEATS, HID);
    scatter_add<<<(N_EDGES * HID + B - 1) / B, B, 0, stream>>>(h1pre, esrc, edst, agg1,
                                                               N_EDGES, HID);
    gemm_wmma_f32src<<<grid2, B, 0, stream>>>(agg1, W2, h2pre,
                                              inv_d, b1, 1,
                                              inv_s, N_NODES, HID, EMB);
  }

  scatter_add<<<(N_EDGES * EMB + B - 1) / B, B, 0, stream>>>(h2pre, esrc, edst, agg2,
                                                             N_EDGES, EMB);
  classify_kernel<<<(N_NODES * 32 + B - 1) / B, B, 0, stream>>>(agg2, inv_d, b2, Wc, bc,
                                                                out, N_NODES);
}
